// SSDLayer_80590766342676
// MI455X (gfx1250) — matrix operand order repaired and checked by reading, not compiled, not run on hardware
//
#include <hip/hip_runtime.h>
#include <hip/hip_bf16.h>

// ---------------------------------------------------------------------------
// SSD layer for MI455X (gfx1250, wave32, WMMA, async-to-LDS).
// B=4, L=4096, DM=2048, H=32, N=128, HD=64.  M = B*L = 16384 tokens.
// Pipeline:
//   0) convert+transpose weights -> bf16 W^T (N x K), zero-padded rows
//   1) GEMM0: Px = x @ Wx                (M x 288)   bf16 WMMA, f32 accum
//   2) GEMM1: G  = sigmoid(x @ Wg + bg)  (M x 2048)
//   3) scan pass1: per-chunk (P = prod a, Q), 16 chunks of 256 steps
//   4) scan pass2: 16-step sequential combine -> chunk start states
//   5) scan pass3: replay chunks, y[m,h] = sum_n h*C (wave32 shfl reduce)
//   6) GEMM2: out = ((y + x*Dp) * G) @ Wo, elementwise fused into A staging
// GEMM: 128x128 block tile, K-stage 64, 8 waves x (32x64) wave tile,
//       B tiles staged via GLOBAL_LOAD_ASYNC_TO_LDS_B128 when available.
// ---------------------------------------------------------------------------

typedef __attribute__((ext_vector_type(16))) __bf16 v16bf;
typedef __attribute__((ext_vector_type(8)))  __bf16 v8bf;
typedef __attribute__((ext_vector_type(8)))  float  v8f;
typedef int v4i_ __attribute__((vector_size(16)));   // matches builtin's V4i

#define DM_   2048
#define H_    32
#define N_    128
#define L_    4096
#define B_    4
#define M_    (B_ * L_)      // 16384
#define NCHUNK 16
#define TC    (L_ / NCHUNK)  // 256
#define KDIM  DM_

#if defined(__has_builtin)
#if __has_builtin(__builtin_amdgcn_global_load_async_to_lds_b128) && \
    __has_builtin(__builtin_amdgcn_s_wait_asynccnt)
#define USE_ASYNC 1
#endif
#endif
#ifndef USE_ASYNC
#define USE_ASYNC 0
#endif

__device__ __forceinline__ v16bf cat16(v8bf lo, v8bf hi) {
  return __builtin_shufflevector(lo, hi, 0,1,2,3,4,5,6,7,8,9,10,11,12,13,14,15);
}

__device__ __forceinline__ float softplus_f(float z) {
  return (z > 20.f) ? z : __logf(1.f + __expf(z));
}

// ---------------------------------------------------------------------------
// Weight convert + transpose: W (K x Nout, fp32) -> WT (NoutPad x K, bf16),
// zero-filling rows Nout..NoutPad-1.  LDS-tiled for coalescing both sides.
// ---------------------------------------------------------------------------
__global__ __launch_bounds__(256)
void convert_transpose_w(const float* __restrict__ W, __bf16* __restrict__ WT,
                         int Nout)
{
  __shared__ float tile[32][33];
  const int k0 = blockIdx.x * 32;
  const int n0 = blockIdx.y * 32;
  const int tx = threadIdx.x & 31;
  const int ty = threadIdx.x >> 5;   // 0..7
  #pragma unroll
  for (int i = 0; i < 4; ++i) {
    int kk = ty + i * 8;
    int n  = n0 + tx;
    tile[kk][tx] = (n < Nout) ? W[(size_t)(k0 + kk) * Nout + n] : 0.f;
  }
  __syncthreads();
  #pragma unroll
  for (int i = 0; i < 4; ++i) {
    int nn = ty + i * 8;
    WT[(size_t)(n0 + nn) * KDIM + k0 + tx] = (__bf16)tile[tx][nn];
  }
}

// ---------------------------------------------------------------------------
// Tiled bf16-WMMA GEMM:  Out(M x Nout) = A(M x 2048) @ WT^T   (WT is N x K)
// MODE 0: plain      MODE 1: sigmoid(v + bias[n])
// MODE 2: A on the fly: A[m,k] = (y[m,k>>6] + x[m,k]*Dp[k>>6]) * G[m,k]
// ---------------------------------------------------------------------------
template <int MODE>
__global__ __launch_bounds__(256)
void gemm_wmma_bf16(const float* __restrict__ Amat,
                    const __bf16* __restrict__ WT,
                    float* __restrict__ Out,
                    const float* __restrict__ bias,
                    const float* __restrict__ Gbuf,
                    const float* __restrict__ Ybuf,
                    const float* __restrict__ Dp,
                    int Nout)
{
  constexpr int LDA = 72;   // bf16 stride: 144B rows -> 16B aligned, no conflicts
  __shared__ __attribute__((aligned(16))) __bf16 As[128][LDA];
  __shared__ __attribute__((aligned(16))) __bf16 Bs[128][LDA];   // [n][k]

  const int tid  = threadIdx.x;
  const int lane = tid & 31;
  const int w    = tid >> 5;
  const int wr   = w >> 1;     // 0..3  (M)
  const int wc   = w & 1;      // 0..1  (N)
  const int m0   = blockIdx.y * 128;
  const int n0   = blockIdx.x * 128;

  const int row16 = lane & 15;
  const int kbA   = (lane >> 4) * 8;    // A frag: K 0-7/16-23 vs 8-15/24-31
  const int kbB   = (lane >> 4) * 8;    // B frag: K 0-7/16-23 vs 8-15/24-31

  v8f acc[2][4] = {};

  for (int k0 = 0; k0 < KDIM; k0 += 64) {
    // ---- B tile: 128(n) x 64(k) bf16, rows contiguous in WT ----
    #pragma unroll
    for (int i = 0; i < 4; ++i) {
      int idx = tid + i * 256;           // 1024 chunks of 16B
      int nn  = idx >> 3;
      int kc  = idx & 7;
      const __bf16* src = WT + (size_t)(n0 + nn) * KDIM + k0 + kc * 8;
      __bf16* dst = &Bs[nn][kc * 8];
#if USE_ASYNC
      __builtin_amdgcn_global_load_async_to_lds_b128(
          (__attribute__((address_space(1))) v4i_*)(unsigned long long)src,
          (__attribute__((address_space(3))) v4i_*)(unsigned)(unsigned long long)dst,
          0, 0);
#else
      *(uint4*)dst = *(const uint4*)src;
#endif
    }

    // ---- A tile: 128(m) x 64(k) fp32 -> bf16 (vectorized, MODE2 fused) ----
    float4 av[8];
    float4 gv[8];
    #pragma unroll
    for (int i = 0; i < 8; ++i) {
      int idx = tid + i * 256;           // 2048 float4 chunks
      int r   = idx >> 4;
      int cc  = idx & 15;
      size_t moff = (size_t)(m0 + r) * KDIM + (k0 + cc * 4);
      av[i] = *(const float4*)(Amat + moff);
      if (MODE == 2) gv[i] = *(const float4*)(Gbuf + moff);
    }
    #pragma unroll
    for (int i = 0; i < 8; ++i) {
      int idx = tid + i * 256;
      int r   = idx >> 4;
      int cc  = idx & 15;
      float4 v = av[i];
      if (MODE == 2) {
        int hh = (k0 + cc * 4) >> 6;     // same head for all 4 lanes of float4
        float yv = Ybuf[(size_t)(m0 + r) * H_ + hh];
        float dp = Dp[hh];
        v.x = (yv + v.x * dp) * gv[i].x;
        v.y = (yv + v.y * dp) * gv[i].y;
        v.z = (yv + v.z * dp) * gv[i].z;
        v.w = (yv + v.w * dp) * gv[i].w;
      }
      union { __bf16 h[4]; uint2 u; } cv;
      cv.h[0] = (__bf16)v.x; cv.h[1] = (__bf16)v.y;
      cv.h[2] = (__bf16)v.z; cv.h[3] = (__bf16)v.w;
      *(uint2*)&As[r][cc * 4] = cv.u;
    }

#if USE_ASYNC
    __builtin_amdgcn_s_wait_asynccnt(0);
#endif
    __syncthreads();

    // ---- 2 k-steps x (2x4) WMMA tiles per wave ----
    #pragma unroll
    for (int ks = 0; ks < 2; ++ks) {
      const int kk0 = ks * 32;
      v16bf afr[2], bfr[4];
      #pragma unroll
      for (int i = 0; i < 2; ++i) {
        const __bf16* ap = &As[wr * 32 + i * 16 + row16][kk0 + kbA];
        afr[i] = cat16(*(const v8bf*)ap, *(const v8bf*)(ap + 16));
      }
      #pragma unroll
      for (int j = 0; j < 4; ++j) {
        const __bf16* bp = &Bs[wc * 64 + j * 16 + row16][kk0 + kbB];
        bfr[j] = cat16(*(const v8bf*)bp, *(const v8bf*)(bp + 16));
      }
      #pragma unroll
      for (int i = 0; i < 2; ++i)
        #pragma unroll
        for (int j = 0; j < 4; ++j)
          acc[i][j] = __builtin_amdgcn_wmma_f32_16x16x32_bf16(
              false, afr[i], false, bfr[j], (short)0, acc[i][j], false, false);
    }
    __syncthreads();
  }

  // ---- epilogue / store (C/D: half-wave = M rows 0-7 / 8-15) ----
  const int mhalf = (lane >= 16) ? 8 : 0;
  #pragma unroll
  for (int i = 0; i < 2; ++i) {
    #pragma unroll
    for (int j = 0; j < 4; ++j) {
      int mb = m0 + wr * 32 + i * 16 + mhalf;
      int nt = n0 + wc * 64 + j * 16 + (lane & 15);
      if (nt < Nout) {
        #pragma unroll
        for (int r = 0; r < 8; ++r) {
          float v = acc[i][j][r];
          if (MODE == 1) { v += bias[nt]; v = 1.f / (1.f + __expf(-v)); }
          Out[(size_t)(mb + r) * Nout + nt] = v;
        }
      }
    }
  }
}

// ---------------------------------------------------------------------------
// Scan pass 1: one wave per (b, h, chunk).  Lane owns 4 n-states (n=lane+32j).
// ---------------------------------------------------------------------------
__global__ __launch_bounds__(128)
void scan_pass1(const float* __restrict__ Px, const float* __restrict__ Aarr,
                float* __restrict__ Pc, float* __restrict__ Qc)
{
  const int lane = threadIdx.x & 31;
  const int task = blockIdx.x * 4 + (threadIdx.x >> 5);   // < B*H*NCHUNK
  const int c  = task & (NCHUNK - 1);
  const int hh = (task >> 4) & (H_ - 1);
  const int b  = task >> 9;

  float Aj[4], Pj[4], Qj[4];
  int nj[4];
  #pragma unroll
  for (int j = 0; j < 4; ++j) {
    nj[j] = lane + 32 * j;
    Aj[j] = Aarr[hh * N_ + nj[j]];
    Pj[j] = 1.f;
    Qj[j] = 0.f;
  }

  const float* base = Px + (size_t)(b * L_ + c * TC) * 288;
  for (int t = 0; t < TC; ++t) {
    float db = base[256 + hh];
    #pragma unroll
    for (int j = 0; j < 4; ++j) {
      float d = softplus_f(db + base[nj[j]]);
      float e = __expf(d * Aj[j]);
      Qj[j] = e * Qj[j] + d * d;
      Pj[j] *= e;
    }
    base += 288;
  }
  #pragma unroll
  for (int j = 0; j < 4; ++j) {
    size_t idx = ((size_t)((b * H_ + hh) * N_ + nj[j])) * NCHUNK + c;
    Pc[idx] = Pj[j];
    Qc[idx] = Qj[j];
  }
}

// ---------------------------------------------------------------------------
// Scan pass 2: sequential combine over 16 chunks, one thread per (b,h,n).
// ---------------------------------------------------------------------------
__global__ void scan_pass2(const float* __restrict__ Pc,
                           const float* __restrict__ Qc,
                           float* __restrict__ Hs)
{
  int i = blockIdx.x * blockDim.x + threadIdx.x;
  if (i >= B_ * H_ * N_) return;
  float run = 0.f;
  #pragma unroll
  for (int c = 0; c < NCHUNK; ++c) {
    Hs[(size_t)i * NCHUNK + c] = run;
    run = Pc[(size_t)i * NCHUNK + c] * run + Qc[(size_t)i * NCHUNK + c];
  }
}

// ---------------------------------------------------------------------------
// Scan pass 3: replay chunk from start state; wave32 shfl reduce over N=128.
// ---------------------------------------------------------------------------
__global__ __launch_bounds__(128)
void scan_pass3(const float* __restrict__ Px, const float* __restrict__ Aarr,
                const float* __restrict__ Hs, float* __restrict__ Y)
{
  const int lane = threadIdx.x & 31;
  const int task = blockIdx.x * 4 + (threadIdx.x >> 5);
  const int c  = task & (NCHUNK - 1);
  const int hh = (task >> 4) & (H_ - 1);
  const int b  = task >> 9;

  float Aj[4], hj[4];
  int nj[4];
  #pragma unroll
  for (int j = 0; j < 4; ++j) {
    nj[j] = lane + 32 * j;
    Aj[j] = Aarr[hh * N_ + nj[j]];
    hj[j] = Hs[((size_t)((b * H_ + hh) * N_ + nj[j])) * NCHUNK + c];
  }

  const int l0 = c * TC;
  const float* base = Px + (size_t)(b * L_ + l0) * 288;
  for (int t = 0; t < TC; ++t) {
    float db = base[256 + hh];
    float part = 0.f;
    #pragma unroll
    for (int j = 0; j < 4; ++j) {
      float d = softplus_f(db + base[nj[j]]);
      float e = __expf(d * Aj[j]);
      hj[j] = e * hj[j] + d * d;
      part += hj[j] * base[128 + nj[j]];   // C_ssm
    }
    #pragma unroll
    for (int off = 16; off > 0; off >>= 1)
      part += __shfl_xor(part, off, 32);
    if (lane == 0)
      Y[(size_t)(b * L_ + l0 + t) * H_ + hh] = part;
    base += 288;
  }
}

// ---------------------------------------------------------------------------
extern "C" void kernel_launch(void* const* d_in, const int* in_sizes, int n_in,
                              void* d_out, int out_size, void* d_ws, size_t ws_size,
                              hipStream_t stream)
{
  const float* x  = (const float*)d_in[0];
  const float* Wx = (const float*)d_in[1];
  const float* Wg = (const float*)d_in[2];
  const float* bg = (const float*)d_in[3];
  const float* Wo = (const float*)d_in[4];
  const float* A  = (const float*)d_in[5];
  const float* Dp = (const float*)d_in[6];
  float* out = (float*)d_out;
  float* ws  = (float*)d_ws;

  // workspace layout
  float* Px = ws;                                 // M * 288
  float* G  = Px + (size_t)M_ * 288;              // M * 2048
  float* Y  = G  + (size_t)M_ * DM_;              // M * 32
  float* Pc = Y  + (size_t)M_ * H_;               // B*H*N*NCHUNK
  float* Qc = Pc + (size_t)B_ * H_ * N_ * NCHUNK;
  float* Hs = Qc + (size_t)B_ * H_ * N_ * NCHUNK;
  __bf16* WxT = (__bf16*)(Hs + (size_t)B_ * H_ * N_ * NCHUNK);  // 384 x K
  __bf16* WgT = WxT + (size_t)384 * KDIM;                       // 2048 x K
  __bf16* WoT = WgT + (size_t)DM_ * KDIM;                       // 2048 x K

  dim3 blk(256);
  // 0) weight convert + transpose (zero-padded)
  convert_transpose_w<<<dim3(KDIM / 32, 384 / 32), blk, 0, stream>>>(Wx, WxT, 288);
  convert_transpose_w<<<dim3(KDIM / 32, DM_ / 32), blk, 0, stream>>>(Wg, WgT, DM_);
  convert_transpose_w<<<dim3(KDIM / 32, DM_ / 32), blk, 0, stream>>>(Wo, WoT, DM_);
  // 1) Px = x @ Wx   (Nout = 288, padded to 384)
  gemm_wmma_bf16<0><<<dim3(384 / 128, M_ / 128), blk, 0, stream>>>(
      x, WxT, Px, nullptr, nullptr, nullptr, nullptr, 288);
  // 2) G = sigmoid(x @ Wg + bg)
  gemm_wmma_bf16<1><<<dim3(DM_ / 128, M_ / 128), blk, 0, stream>>>(
      x, WgT, G, bg, nullptr, nullptr, nullptr, DM_);
  // 3-5) chunked scan
  scan_pass1<<<(B_ * H_ * NCHUNK) / 4, 128, 0, stream>>>(Px, A, Pc, Qc);
  scan_pass2<<<(B_ * H_ * N_ + 255) / 256, 256, 0, stream>>>(Pc, Qc, Hs);
  scan_pass3<<<(B_ * H_ * NCHUNK) / 4, 128, 0, stream>>>(Px, A, Hs, Y);
  // 6) out = ((Y + x*Dp) * G) @ Wo, fused into A staging
  gemm_wmma_bf16<2><<<dim3(DM_ / 128, M_ / 128), blk, 0, stream>>>(
      x, WoT, out, nullptr, G, Y, Dp, DM_);
}
